// OT_Attn_assem_50113678410551
// MI455X (gfx1250) — matrix operand order repaired and checked by reading, compiled, not run on hardware
//
#include <hip/hip_runtime.h>
#include <hip/hip_bf16.h>

#define N_SRC 8192
#define M_TGT 4096
#define DIM   256
#define OT_REG 0.1f
#define OT_FI  (0.5f / (0.5f + 0.1f))   // tau/(tau+reg) = 0.833333
#define N_SINK 250

typedef _Float16 h2   __attribute__((ext_vector_type(2)));
typedef _Float16 h8   __attribute__((ext_vector_type(8)));
typedef _Float16 v16h __attribute__((ext_vector_type(16)));
typedef float    v8f  __attribute__((ext_vector_type(8)));

// ---------------------------------------------------------------------------
// Row normalize: x - mean, then L2-normalize. Writes f32 and f16 copies.
// One block (256 threads) per row of length 256.
// ---------------------------------------------------------------------------
__global__ void normalize_kernel(const float* __restrict__ x,
                                 float* __restrict__ yf,
                                 _Float16* __restrict__ yh) {
    __shared__ float red[256];
    const int row = blockIdx.x;
    const int c = threadIdx.x;
    const size_t idx = (size_t)row * DIM + c;
    float val = x[idx];
    red[c] = val; __syncthreads();
    #pragma unroll
    for (int s = 128; s; s >>= 1) { if (c < s) red[c] += red[c + s]; __syncthreads(); }
    const float mean = red[0] * (1.0f / DIM);
    __syncthreads();
    const float y = val - mean;
    red[c] = y * y; __syncthreads();
    #pragma unroll
    for (int s = 128; s; s >>= 1) { if (c < s) red[c] += red[c + s]; __syncthreads(); }
    const float inv = 1.0f / fmaxf(sqrtf(red[0]), 1e-8f);
    const float out = y * inv;
    yf[idx] = out;
    yh[idx] = (_Float16)out;
}

// ---------------------------------------------------------------------------
// LDS-tiled transpose tgt16 (4096x256) -> tgtT16 (256x4096). 32x32 tiles,
// padded LDS row to avoid bank conflicts; coalesced reads AND writes.
// ---------------------------------------------------------------------------
__global__ void transpose_kernel(const _Float16* __restrict__ in,
                                 _Float16* __restrict__ out) {
    __shared__ _Float16 tile[32][33];
    const int bx = blockIdx.x;          // 0..7   (over DIM)
    const int by = blockIdx.y;          // 0..127 (over M_TGT)
    const int tx = threadIdx.x & 31;
    const int ty = threadIdx.x >> 5;    // 0..7
    #pragma unroll
    for (int r = 0; r < 32; r += 8)
        tile[ty + r][tx] = in[(size_t)(by * 32 + ty + r) * DIM + bx * 32 + tx];
    __syncthreads();
    #pragma unroll
    for (int r = 0; r < 32; r += 8)
        out[(size_t)(bx * 32 + ty + r) * M_TGT + by * 32 + tx] = tile[tx][ty + r];
}

// ---------------------------------------------------------------------------
// GEMM1: cost = max(2 - 2 * srcN @ tgtNᵀ, 0), stored f16; per-block max out.
// One wave32 per 16x16 output tile, K loop of 8x v_wmma_f32_16x16x32_f16.
// A-lane layout: row = lane%16, K chunks [h*8, h*8+8) and [16+h*8, ...).
// B-lane layout: col = lane%16 (a row of tgt), K chunks [h*8, h*8+8) and [16+h*8, ...).
// ---------------------------------------------------------------------------
__global__ void gemm_cost_kernel(const _Float16* __restrict__ src16,
                                 const _Float16* __restrict__ tgt16,
                                 _Float16* __restrict__ cost16,
                                 float* __restrict__ blockmax) {
    __shared__ float red[256];
    const int lane = threadIdx.x & 31;
    const int wid  = blockIdx.x * 8 + (threadIdx.x >> 5);
    const int tile_m = wid >> 8;    // 512 tiles over N_SRC
    const int tile_n = wid & 255;   // 256 tiles over M_TGT
    const int hh  = lane >> 4;
    const int l16 = lane & 15;
    const _Float16* arow = src16 + (size_t)(tile_m * 16 + l16) * DIM;
    const _Float16* brow = tgt16 + (size_t)(tile_n * 16 + l16) * DIM;
    v8f acc = {0.f, 0.f, 0.f, 0.f, 0.f, 0.f, 0.f, 0.f};
    #pragma unroll
    for (int k0 = 0; k0 < DIM; k0 += 32) {
        h8 a0 = *(const h8*)(arow + k0 + hh * 8);
        h8 a1 = *(const h8*)(arow + k0 + 16 + hh * 8);
        h8 b0 = *(const h8*)(brow + k0 + hh * 8);
        h8 b1 = *(const h8*)(brow + k0 + 16 + hh * 8);
        v16h A, B;
        #pragma unroll
        for (int e = 0; e < 8; ++e) { A[e] = a0[e]; A[e + 8] = a1[e]; B[e] = b0[e]; B[e + 8] = b1[e]; }
        acc = __builtin_amdgcn_wmma_f32_16x16x32_f16(false, A, false, B, (short)0, acc, false, false);
    }
    float lmax = 0.f;
    _Float16* cbase = cost16 + (size_t)(tile_m * 16 + hh * 8) * M_TGT + tile_n * 16 + l16;
    #pragma unroll
    for (int r = 0; r < 8; ++r) {
        float c = fmaxf(2.0f - 2.0f * acc[r], 0.0f);
        lmax = fmaxf(lmax, c);
        cbase[(size_t)r * M_TGT] = (_Float16)c;
    }
    red[threadIdx.x] = lmax; __syncthreads();
    #pragma unroll
    for (int s = 128; s; s >>= 1) {
        if (threadIdx.x < s) red[threadIdx.x] = fmaxf(red[threadIdx.x], red[threadIdx.x + s]);
        __syncthreads();
    }
    if (threadIdx.x == 0) blockmax[blockIdx.x] = red[0];
}

__global__ void reduce_max_kernel(const float* __restrict__ p, int n, float* __restrict__ out) {
    __shared__ float red[1024];
    float m = 0.f;
    for (int i = threadIdx.x; i < n; i += 1024) m = fmaxf(m, p[i]);
    red[threadIdx.x] = m; __syncthreads();
    #pragma unroll
    for (int s = 512; s; s >>= 1) {
        if (threadIdx.x < s) red[threadIdx.x] = fmaxf(red[threadIdx.x], red[threadIdx.x + s]);
        __syncthreads();
    }
    if (threadIdx.x == 0) out[0] = red[0];
}

// K = exp(-cost / (cmax * REG)), packed 2xf16 (K stays L2-resident: 67MB < 192MB)
__global__ void gibbs_kernel(const _Float16* __restrict__ cost16,
                             const float* __restrict__ cmax,
                             _Float16* __restrict__ Kk) {
    const float scale = 1.0f / (cmax[0] * OT_REG);
    const int total2 = (N_SRC * M_TGT) / 2;
    for (int i = blockIdx.x * blockDim.x + threadIdx.x; i < total2; i += gridDim.x * blockDim.x) {
        h2 c = *(const h2*)(cost16 + 2 * (size_t)i);
        h2 k;
        k[0] = (_Float16)__expf(-(float)c[0] * scale);
        k[1] = (_Float16)__expf(-(float)c[1] * scale);
        *(h2*)(Kk + 2 * (size_t)i) = k;
    }
}

__global__ void init_uv_kernel(float* __restrict__ u, float* __restrict__ v) {
    const int i = blockIdx.x * blockDim.x + threadIdx.x;
    if (i < N_SRC) u[i] = 1.0f;
    if (i < M_TGT) v[i] = 1.0f;
}

// u = (a / (K @ v))^fi : one wave32 per row, 16B packed f16 loads, shuffle reduce
__global__ void row_update_kernel(const _Float16* __restrict__ Kk,
                                  const float* __restrict__ v,
                                  float* __restrict__ u) {
    const int lane = threadIdx.x & 31;
    const int row = (blockIdx.x * blockDim.x + threadIdx.x) >> 5;
    const _Float16* rowp = Kk + (size_t)row * M_TGT;
    float s = 0.f;
    #pragma unroll 4
    for (int j0 = lane * 8; j0 < M_TGT; j0 += 256) {
        h8 kk = *(const h8*)(rowp + j0);
        const float* vp = v + j0;
        #pragma unroll
        for (int e = 0; e < 8; ++e) s += (float)kk[e] * vp[e];
    }
    #pragma unroll
    for (int off = 16; off; off >>= 1) s += __shfl_xor(s, off, 32);
    if (lane == 0) u[row] = __powf((1.0f / N_SRC) / s, OT_FI);
}

// Kᵀ @ u, 2 columns per thread (packed 4B loads -> full 128B/wave), split over
// 8 row-chunks for parallelism; deterministic partials.
__global__ void col_part_kernel(const _Float16* __restrict__ Kk,
                                const float* __restrict__ u,
                                float* __restrict__ cpart) {
    const int col = (blockIdx.x * blockDim.x + threadIdx.x) * 2;   // 0,2,..,4094
    const int r0 = blockIdx.y * (N_SRC / 8);
    const _Float16* p = Kk + (size_t)r0 * M_TGT + col;
    float s0 = 0.f, s1 = 0.f;
    #pragma unroll 4
    for (int i = 0; i < N_SRC / 8; ++i) {
        h2 kk = *(const h2*)p;
        const float ui = u[r0 + i];
        s0 += (float)kk[0] * ui;
        s1 += (float)kk[1] * ui;
        p += M_TGT;
    }
    cpart[blockIdx.y * M_TGT + col]     = s0;
    cpart[blockIdx.y * M_TGT + col + 1] = s1;
}

__global__ void col_finish_kernel(const float* __restrict__ cpart, float* __restrict__ v) {
    const int col = blockIdx.x * blockDim.x + threadIdx.x;
    float s = 0.f;
    #pragma unroll
    for (int k = 0; k < 8; ++k) s += cpart[k * M_TGT + col];
    v[col] = __powf((1.0f / M_TGT) / s, OT_FI);
}

// ---------------------------------------------------------------------------
// GEMM2: fused = srcN + (u ∘ K ∘ v) @ tgt. flow built on the fly in f16,
// scaled by 8192 to dodge f16 underflow (entries ~1e-8..1e-4), rescaled after.
// v staged in LDS (16KB) once per block. One wave32 per 16x16 tile;
// 128 x v_wmma_f32_16x16x32_f16 over K=4096.
// ---------------------------------------------------------------------------
__global__ void gemm_fused_kernel(const _Float16* __restrict__ Kk,
                                  const _Float16* __restrict__ tgtT16,
                                  const float* __restrict__ srcN,
                                  const float* __restrict__ u,
                                  const float* __restrict__ v,
                                  float* __restrict__ out) {
    __shared__ float vlds[M_TGT];
    for (int i = threadIdx.x; i < M_TGT; i += 256) vlds[i] = v[i];
    __syncthreads();

    const int lane = threadIdx.x & 31;
    const int wid = blockIdx.x * 8 + (threadIdx.x >> 5);
    const int tile_m = wid >> 4;   // 512 tiles over N_SRC
    const int tile_n = wid & 15;   // 16 tiles over DIM
    const int hh  = lane >> 4;
    const int l16 = lane & 15;
    const int m = tile_m * 16 + l16;
    const float us = u[m] * 8192.0f;
    const _Float16* krow = Kk + (size_t)m * M_TGT;
    const _Float16* brow = tgtT16 + (size_t)(tile_n * 16 + l16) * M_TGT;
    v8f acc = {0.f, 0.f, 0.f, 0.f, 0.f, 0.f, 0.f, 0.f};
    for (int k0 = 0; k0 < M_TGT; k0 += 32) {
        const int ka = k0 + hh * 8;
        const int kb = k0 + 16 + hh * 8;
        h8 KA = *(const h8*)(krow + ka);
        h8 KB = *(const h8*)(krow + kb);
        const float* va = vlds + ka;
        const float* vb = vlds + kb;
        v16h A;
        #pragma unroll
        for (int e = 0; e < 8; ++e) {
            A[e]     = (_Float16)(us * (float)KA[e] * va[e]);
            A[e + 8] = (_Float16)(us * (float)KB[e] * vb[e]);
        }
        h8 b0 = *(const h8*)(brow + k0 + hh * 8);
        h8 b1 = *(const h8*)(brow + k0 + 16 + hh * 8);
        v16h B;
        #pragma unroll
        for (int e = 0; e < 8; ++e) { B[e] = b0[e]; B[e + 8] = b1[e]; }
        acc = __builtin_amdgcn_wmma_f32_16x16x32_f16(false, A, false, B, (short)0, acc, false, false);
    }
    #pragma unroll
    for (int r = 0; r < 8; ++r) {
        const int mr = tile_m * 16 + hh * 8 + r;
        const size_t o = (size_t)mr * DIM + tile_n * 16 + l16;
        out[o] = srcN[o] + acc[r] * (1.0f / 8192.0f);
    }
}

// dist = Σ cost ∘ (u ∘ K ∘ v), two-pass deterministic reduction
__global__ void dist_part_kernel(const _Float16* __restrict__ cost16,
                                 const _Float16* __restrict__ Kk,
                                 const float* __restrict__ u,
                                 const float* __restrict__ v,
                                 float* __restrict__ part) {
    __shared__ float red[256];
    float s = 0.f;
    const int total = N_SRC * M_TGT;
    for (int idx = blockIdx.x * blockDim.x + threadIdx.x; idx < total; idx += gridDim.x * blockDim.x) {
        const int i = idx >> 12;
        const int j = idx & (M_TGT - 1);
        s += (float)cost16[idx] * u[i] * (float)Kk[idx] * v[j];
    }
    red[threadIdx.x] = s; __syncthreads();
    #pragma unroll
    for (int t = 128; t; t >>= 1) {
        if (threadIdx.x < t) red[threadIdx.x] += red[threadIdx.x + t];
        __syncthreads();
    }
    if (threadIdx.x == 0) part[blockIdx.x] = red[0];
}

__global__ void dist_finish_kernel(const float* __restrict__ part, float* __restrict__ out) {
    __shared__ float red[256];
    float s = 0.f;
    for (int i = threadIdx.x; i < 1024; i += 256) s += part[i];
    red[threadIdx.x] = s; __syncthreads();
    #pragma unroll
    for (int t = 128; t; t >>= 1) {
        if (threadIdx.x < t) red[threadIdx.x] += red[threadIdx.x + t];
        __syncthreads();
    }
    if (threadIdx.x == 0) out[0] = red[0];
}

// ---------------------------------------------------------------------------
extern "C" void kernel_launch(void* const* d_in, const int* in_sizes, int n_in,
                              void* d_out, int out_size, void* d_ws, size_t ws_size,
                              hipStream_t stream) {
    (void)in_sizes; (void)n_in; (void)out_size; (void)ws_size;
    const float* shared_f   = (const float*)d_in[0];   // 8192 x 256
    const float* semantic_f = (const float*)d_in[1];   // 4096 x 256

    char* ws = (char*)d_ws;
    float*    srcN   = (float*)   (ws + 0);           //  8 MB
    float*    tgtN   = (float*)   (ws + 8388608);     //  4 MB (f32 normalized tgt)
    _Float16* src16  = (_Float16*)(ws + 12582912);    //  4 MB
    _Float16* tgt16  = (_Float16*)(ws + 16777216);    //  2 MB
    _Float16* tgtT16 = (_Float16*)(ws + 18874368);    //  2 MB (256 x 4096)
    _Float16* cost16 = (_Float16*)(ws + 20971520);    // 64 MB
    _Float16* K16    = (_Float16*)(ws + 88080384);    // 64 MB
    float*    u      = (float*)   (ws + 155189248);   // 32 KB
    float*    v      = (float*)   (ws + 155222016);   // 16 KB
    float*    cpart  = (float*)   (ws + 155238400);   // 128 KB (8 x 4096)
    float*    parts  = (float*)   (ws + 155369472);   // 64 KB (16384 floats)
    float*    cmax   = (float*)   (ws + 155435008);   // scalar

    float* fused = (float*)d_out;                     // 8192 x 256
    float* dist  = fused + (size_t)N_SRC * DIM;       // scalar (last element)

    normalize_kernel<<<N_SRC, 256, 0, stream>>>(shared_f, srcN, src16);
    normalize_kernel<<<M_TGT, 256, 0, stream>>>(semantic_f, tgtN, tgt16);
    transpose_kernel<<<dim3(DIM / 32, M_TGT / 32), 256, 0, stream>>>(tgt16, tgtT16);

    gemm_cost_kernel<<<16384, 256, 0, stream>>>(src16, tgt16, cost16, parts);
    reduce_max_kernel<<<1, 1024, 0, stream>>>(parts, 16384, cmax);
    gibbs_kernel<<<8192, 256, 0, stream>>>(cost16, cmax, K16);

    init_uv_kernel<<<32, 256, 0, stream>>>(u, v);
    for (int it = 0; it < N_SINK; ++it) {
        row_update_kernel<<<1024, 256, 0, stream>>>(K16, v, u);
        col_part_kernel<<<dim3(8, 8), 256, 0, stream>>>(K16, u, cpart);
        col_finish_kernel<<<16, 256, 0, stream>>>(cpart, v);
    }

    gemm_fused_kernel<<<1024, 256, 0, stream>>>(K16, tgtT16, srcN, u, v, fused);
    dist_part_kernel<<<1024, 256, 0, stream>>>(cost16, K16, u, v, parts);
    dist_finish_kernel<<<1, 256, 0, stream>>>(parts, dist);
}
